// My_RNN_13726715478291
// MI455X (gfx1250) — matrix operand order repaired and checked by reading, not compiled, not run on hardware
//
#include <hip/hip_runtime.h>
#include <cmath>

// Problem constants (match reference setup_inputs)
#define B_  64
#define T_  2048
#define D_  256
#define H_  256
#define K_  512            // fused K = H (recurrent, W1) + D (input, W2)
#define NKT 16             // K_/32 k-tiles per WMMA chain
#define BM  16             // batch rows per workgroup (one WMMA M-tile)

typedef __bf16 v16bf __attribute__((ext_vector_type(16)));
typedef float  v8f   __attribute__((ext_vector_type(8)));

union Frag32B { v16bf v; uint4 q[2]; };

__device__ __forceinline__ unsigned short f2bf(float f) {
  unsigned int u = __builtin_bit_cast(unsigned int, f);
  u += 0x7fffu + ((u >> 16) & 1u);      // round-to-nearest-even
  return (unsigned short)(u >> 16);
}
__device__ __forceinline__ unsigned int pack2bf(float lo, float hi) {
  return (unsigned int)f2bf(lo) | ((unsigned int)f2bf(hi) << 16);
}

// ---------------------------------------------------------------------------
// Prep: convert Wc = [W1 ; W2] (f32, row-major [K][N]) into bf16 fragments in
// WMMA B-matrix (32x16 bf16) per-lane order with contiguous k = 16h + e (as A), so the scan kernel reads B
// with contiguous 32B LDS loads.
// wfrag[((nt*NKT + kt)*32 + lane)*16 + e]  <->  K = kt*32 + (lane<16?0:16)+e,
//                                               N = nt*16 + (lane&15)
// ---------------------------------------------------------------------------
__global__ __launch_bounds__(256) void rnn_prep_wfrag(
    const float* __restrict__ W1, const float* __restrict__ W2,
    unsigned short* __restrict__ wfrag) {
  int tid  = blockIdx.x * 256 + threadIdx.x;   // 0 .. K_*H_-1 (131072)
  int e    = tid & 15;
  int lane = (tid >> 4) & 31;
  int kt   = (tid >> 9) & 15;
  int nt   = tid >> 13;
  int K = kt * 32 + ((lane & 16) ? 16 : 0) + e;
  int N = nt * 16 + (lane & 15);
  float w = (K < H_) ? W1[K * H_ + N] : W2[(K - H_) * H_ + N];
  wfrag[tid] = f2bf(w);
}

// ---------------------------------------------------------------------------
// Fused scan: per workgroup, 16 batch rows. LDS holds Wc (bf16, 256KB, resident)
// and A = [h | x_t] (16x512 bf16). Each of 8 waves owns 2 N-tiles; per step:
// 16 k-tiles x 2 v_wmma_f32_16x16x32_bf16 with f32 accumulation, + bias + tanh,
// h written back to LDS. x_{t+1} is prefetched into registers during compute.
// ---------------------------------------------------------------------------
__global__ __launch_bounds__(256) void rnn_scan(
    const float* __restrict__ x, const float* __restrict__ bias,
    const unsigned short* __restrict__ wfrag, float* __restrict__ out) {
  extern __shared__ unsigned short smem[];
  unsigned short* sW = smem;                 // 512*256 ushorts = 256 KB
  unsigned short* sA = smem + K_ * H_;       // 16*512 ushorts  = 16 KB

  const int tid  = threadIdx.x;
  const int lane = tid & 31;
  const int wave = tid >> 5;
  const int b0   = blockIdx.x * BM;

  // One-time: pull fragment-swizzled W into LDS (16384 b128 transfers / WG).
  {
    const uint4* g = (const uint4*)wfrag;
    uint4* l = (uint4*)sW;
    #pragma unroll 4
    for (int i = tid; i < (K_ * H_ * 2) / 16; i += 256) l[i] = g[i];
  }
  // Zero h region of A (k < H_): h_0 = 0.
  for (int i = tid; i < BM * H_ / 2; i += 256) {
    int m = i >> 7;                 // 128 uints per row
    int c = i & 127;
    ((unsigned int*)(sA + m * K_))[c] = 0u;
  }

  const int nt0  = wave * 2, nt1 = wave * 2 + 1;
  const int ncol = lane & 15;
  const float bv0 = bias[nt0 * 16 + ncol];
  const float bv1 = bias[nt1 * 16 + ncol];
  const int koff  = (lane < 16) ? 0 : 16;  // A-frag K-half per lane (contiguous k = 16h + s)
  const int mbase = (lane < 16) ? 0 : 8;   // ISA f32 C/D M-half per lane

  // Per-thread x staging: row m, 16 contiguous floats starting at d0.
  const int xm = tid >> 4;
  const int xd = (tid & 15) << 4;
  const float* xrow = x + ((size_t)(b0 + xm) * T_) * D_ + xd;
  unsigned int* xdst = (unsigned int*)(sA + xm * K_ + H_ + xd);

  float4 r0, r1, r2, r3;
  {
    const float4* s = (const float4*)xrow;               // t = 0
    r0 = s[0]; r1 = s[1]; r2 = s[2]; r3 = s[3];
  }

  for (int t = 0; t < T_; ++t) {
    // (a) stage x_t (prefetched last iter) into A's input half, as bf16
    xdst[0] = pack2bf(r0.x, r0.y); xdst[1] = pack2bf(r0.z, r0.w);
    xdst[2] = pack2bf(r1.x, r1.y); xdst[3] = pack2bf(r1.z, r1.w);
    xdst[4] = pack2bf(r2.x, r2.y); xdst[5] = pack2bf(r2.z, r2.w);
    xdst[6] = pack2bf(r3.x, r3.y); xdst[7] = pack2bf(r3.z, r3.w);
    __syncthreads();                 // A = [h_{t-1} | x_t] fully visible

    if (t + 1 < T_) {                // (b) prefetch x_{t+1}; overlaps WMMA chain
      const float4* s = (const float4*)(xrow + (size_t)(t + 1) * D_);
      r0 = s[0]; r1 = s[1]; r2 = s[2]; r3 = s[3];
    }

    // (c) acc[16x16] += A[16x512] * Wc[512x16] for the wave's two N-tiles
    v8f acc0 = {}; v8f acc1 = {};
    const unsigned short* Arow = sA + (lane & 15) * K_;
    #pragma unroll
    for (int kt = 0; kt < NKT; ++kt) {
      Frag32B a;
      a.q[0] = *(const uint4*)(Arow + kt * 32 + koff);
      a.q[1] = *(const uint4*)(Arow + kt * 32 + koff + 8);
      Frag32B bA, bB;
      const uint4* p0 = (const uint4*)(sW + (((nt0 * NKT + kt) * 32 + lane) << 4));
      bA.q[0] = p0[0]; bA.q[1] = p0[1];
      const uint4* p1 = (const uint4*)(sW + (((nt1 * NKT + kt) * 32 + lane) << 4));
      bB.q[0] = p1[0]; bB.q[1] = p1[1];
      acc0 = __builtin_amdgcn_wmma_f32_16x16x32_bf16(false, a.v, false, bA.v,
                                                     (short)0, acc0, false, false);
      acc1 = __builtin_amdgcn_wmma_f32_16x16x32_bf16(false, a.v, false, bB.v,
                                                     (short)0, acc1, false, false);
    }
    __syncthreads();                 // all reads of A complete before h rewrite

    // (d) h_t = tanh(acc + b); requantize into A's h half; emit f32 at t=T-1
    #pragma unroll
    for (int r = 0; r < 8; ++r) {
      float h0 = tanhf(acc0[r] + bv0);
      float h1 = tanhf(acc1[r] + bv1);
      int m = mbase + r;
      sA[m * K_ + nt0 * 16 + ncol] = f2bf(h0);
      sA[m * K_ + nt1 * 16 + ncol] = f2bf(h1);
      if (t == T_ - 1) {
        out[(size_t)(b0 + m) * H_ + nt0 * 16 + ncol] = h0;
        out[(size_t)(b0 + m) * H_ + nt1 * 16 + ncol] = h1;
      }
    }
  }
}

// ---------------------------------------------------------------------------
extern "C" void kernel_launch(void* const* d_in, const int* in_sizes, int n_in,
                              void* d_out, int out_size, void* d_ws, size_t ws_size,
                              hipStream_t stream) {
  const float* x    = (const float*)d_in[0];   // [64, 2048, 256] f32
  const float* W1   = (const float*)d_in[1];   // [256, 256] f32
  const float* W2   = (const float*)d_in[2];   // [256, 256] f32
  const float* bias = (const float*)d_in[3];   // [256] f32
  float* out        = (float*)d_out;           // [64, 256] f32
  unsigned short* wfrag = (unsigned short*)d_ws;  // 256 KB of workspace

  // 1) swizzle [W1;W2] -> bf16 WMMA B-fragments in d_ws
  rnn_prep_wfrag<<<(K_ * H_) / 256, 256, 0, stream>>>(W1, W2, wfrag);

  // 2) fused projection+scan; 272 KB dynamic LDS (fits CDNA5's 320 KB/WGP)
  size_t lds_bytes = (size_t)(K_ * H_ + BM * K_) * sizeof(unsigned short);
  (void)hipFuncSetAttribute(reinterpret_cast<const void*>(rnn_scan),
                            hipFuncAttributeMaxDynamicSharedMemorySize,
                            (int)lds_bytes);
  rnn_scan<<<B_ / BM, 256, lds_bytes, stream>>>(x, bias, wfrag, out);
}
